// ModularRIN_75892072120993
// MI455X (gfx1250) — compile-verified, matrix operand order checked by reading, not run on hardware
//
#include <hip/hip_runtime.h>
#include <hip/hip_bf16.h>
#include <math.h>

// ---------------------------------------------------------------------------
// ModularRIN forward for MI455X (gfx1250, wave32, WMMA).
// The scan carry is (h_real,h_imag) only -> layer matmuls for steps 0..62 are
// dead code. Run the elementwise rotation recurrence for 64 steps, then the
// two resonant layers + output projection ONCE.
// GEMMs: v_wmma_f32_16x16x32_f16, A panel staged in LDS (shared by 8 waves
// per block), B weights streamed f32->f16 inline (each element read once).
// ---------------------------------------------------------------------------

#define PHI 1.6180339887498949f

#define BB 128
#define SS 64
#define VV 1024
#define DD 512
#define NN 1024
#define LL 2

typedef _Float16 half_t;
typedef __attribute__((ext_vector_type(16))) _Float16 v16h;
typedef __attribute__((ext_vector_type(8)))  _Float16 v8h;
typedef __attribute__((ext_vector_type(8)))  float    v8f;

// LDS row padding: +8 halves (16B) so fragment loads across 16 rows hit
// distinct 16B bank groups (row stride 1040B = 16B mod 256B).
#define PAD 8

// ---------------------------------------------------------------------------
// Fragment loaders per CDNA5 ISA 7.12.2 (wave32, 16-bit A 16x32 MxK):
// lanes 0-15: row M=lane, halves 0..7 = K 0..7, halves 8..15 = K 16..23;
// lanes 16-31: row M=lane-16, K offsets +8 / +24.
// ---------------------------------------------------------------------------
__device__ __forceinline__ v16h frag_lds(const half_t* sA, int ld, int k0) {
    const int lane = threadIdx.x & 31;
    const half_t* p = sA + (lane & 15) * ld + k0 + ((lane >> 4) << 3);
    v8h lo = *(const v8h*)(p);
    v8h hi = *(const v8h*)(p + 16);
    v16h f;
#pragma unroll
    for (int i = 0; i < 8; ++i) { f[i] = lo[i]; f[8 + i] = hi[i]; }
    return f;
}

// Load fp32 weights, convert to f16 fragment on the fly.
__device__ __forceinline__ v16h frag_f32(const float* base, int ld, int row0, int k0) {
    const int lane = threadIdx.x & 31;
    const float* p = base + (size_t)(row0 + (lane & 15)) * ld + k0 + ((lane >> 4) << 3);
    v16h f;
#pragma unroll
    for (int i = 0; i < 8; ++i) { f[i] = (half_t)p[i]; f[8 + i] = (half_t)p[16 + i]; }
    return f;
}

// f32 C/D 16x16: VGPR v, lanes 0-15 -> (M=v, N=lane); lanes 16-31 -> (M=8+v).
__device__ __forceinline__ int cd_row(int v) { return (((threadIdx.x & 31) >> 4) << 3) + v; }
__device__ __forceinline__ int cd_col()      { return threadIdx.x & 15; }

#define WMMA_F16(A, B, C) \
    __builtin_amdgcn_wmma_f32_16x16x32_f16(false, (A), false, (B), (short)0, (C), false, false)

// Cooperative copy of a 16-row f16 panel (row-major, ld_src halves) into LDS
// with padded rows. 256 threads, 16B chunks.
__device__ __forceinline__ void stage_panel(half_t* sDst, const half_t* gSrc,
                                            int m0, int ld_src, int kfull) {
    const int chunks = 16 * (kfull >> 3);
    for (int i = threadIdx.x; i < chunks; i += 256) {
        const int r = i / (kfull >> 3);
        const int c = (i % (kfull >> 3)) << 3;
        *(v8h*)&sDst[r * (kfull + PAD) + c] =
            *(const v8h*)&gSrc[(size_t)(m0 + r) * ld_src + c];
    }
}

// ---------------------------------------------------------------------------
// Kernel 1: rotation recurrence. One thread per (b, d) element; 64 steps.
// ---------------------------------------------------------------------------
__global__ void rin_scan(const int* __restrict__ ids,
                         const float* __restrict__ emb,   // (V, 2D)
                         float* __restrict__ h_f, half_t* __restrict__ h_h) {
    const int idx = blockIdx.x * blockDim.x + threadIdx.x;   // 0 .. B*D-1
    const int b = idx >> 9;          // / DD
    const int dd = idx & (DD - 1);
    float hr = 0.0f, hi = 0.0f;
#pragma unroll 4
    for (int s = 0; s < SS; ++s) {
        const int id = ids[b * SS + s];                      // uniform over dd
        const float* e = emb + (size_t)id * (2 * DD);
        const float w  = e[dd];
        const float bt = e[DD + dd];
        const float wl = 1.0f + fabsf(w);
        const float tv = (float)s * PHI;
        const float tr = hr / wl + bt + tv;
        const float ti = hi / wl + bt + tv;
        float sr, cr, si, ci;
        __sincosf(tr, &sr, &cr);
        __sincosf(ti, &si, &ci);
        hr = cr * ci - sr * si;
        hi = cr * si + sr * ci;
    }
    const float h = hr + hi;
    h_f[idx] = h;
    h_h[idx] = (half_t)h;
}

// ---------------------------------------------------------------------------
// Kernel 2: theta = h @ W^T + bias + t63 ; emit cos/sin(theta) as f16.
// M=128, N=1024, K=512. Block = 256 thr = 8 waves; block tile 16 x 128;
// one 16x16 tile per wave. A panel (16 x 512 f16 = 16KB) staged in LDS.
// Grid (N/128, M/16) = (8, 8) -> 512 waves.
// ---------------------------------------------------------------------------
__global__ void rin_theta(const half_t* __restrict__ hA,   // (B, D) f16
                          const float* __restrict__ W,     // (N, D) f32
                          const float* __restrict__ bias,  // (N,)
                          half_t* __restrict__ ct, half_t* __restrict__ st,
                          float tval) {
    __shared__ half_t sA[16 * (DD + PAD)];
    const int m0 = blockIdx.y << 4;
    const int n0 = (blockIdx.x << 7) + ((threadIdx.x >> 5) << 4);

    stage_panel(sA, hA, m0, DD, DD);
    __syncthreads();

    v8f acc = {};
#pragma unroll 4
    for (int k = 0; k < DD; k += 32) {
        __builtin_prefetch(W + (size_t)n0 * DD + k + 128, 0, 0);
        v16h a = frag_lds(sA, DD + PAD, k);
        v16h b = frag_f32(W, DD, n0, k);
        acc = WMMA_F16(a, b, acc);
    }

    const int n = n0 + cd_col();
    const float bi = bias[n] + tval;
#pragma unroll
    for (int v = 0; v < 8; ++v) {
        const int m = m0 + cd_row(v);
        float s_, c_;
        __sincosf(acc[v] + bi, &s_, &c_);
        ct[(size_t)m * NN + n] = (half_t)c_;
        st[(size_t)m * NN + n] = (half_t)s_;
    }
}

// ---------------------------------------------------------------------------
// Kernel 3: out = ct @ Pr^T + st @ Pi^T ; h = h + silu(out).
// M=128, N=512, K=1024. Block tile 16 x 128; cos/sin panels (2 x 33KB) in LDS.
// Grid (N/128, M/16) = (4, 8) -> 256 waves.
// ---------------------------------------------------------------------------
__global__ void rin_proj(const half_t* __restrict__ ct, const half_t* __restrict__ st,
                         const float* __restrict__ Pr,  // (D, N) f32
                         const float* __restrict__ Pi,  // (D, N) f32
                         float* __restrict__ h_f, half_t* __restrict__ h_h) {
    __shared__ half_t sCT[16 * (NN + PAD)];
    __shared__ half_t sST[16 * (NN + PAD)];
    const int m0 = blockIdx.y << 4;
    const int n0 = (blockIdx.x << 7) + ((threadIdx.x >> 5) << 4);

    stage_panel(sCT, ct, m0, NN, NN);
    stage_panel(sST, st, m0, NN, NN);
    __syncthreads();

    v8f acc = {};
#pragma unroll 2
    for (int k = 0; k < NN; k += 32) {
        __builtin_prefetch(Pr + (size_t)n0 * NN + k + 128, 0, 0);
        __builtin_prefetch(Pi + (size_t)n0 * NN + k + 128, 0, 0);
        v16h ac = frag_lds(sCT, NN + PAD, k);
        v16h br = frag_f32(Pr, NN, n0, k);
        acc = WMMA_F16(ac, br, acc);
        v16h as = frag_lds(sST, NN + PAD, k);
        v16h bi = frag_f32(Pi, NN, n0, k);
        acc = WMMA_F16(as, bi, acc);
    }

    const int n = n0 + cd_col();
#pragma unroll
    for (int v = 0; v < 8; ++v) {
        const int m = m0 + cd_row(v);
        const float o = acc[v];
        const float sl = o / (1.0f + __expf(-o));            // silu
        const float hn = h_f[(size_t)m * DD + n] + sl;
        h_f[(size_t)m * DD + n] = hn;
        h_h[(size_t)m * DD + n] = (half_t)hn;
    }
}

// ---------------------------------------------------------------------------
// Kernel 4: logits = h @ output_proj^T. M=128, N=1024(V), K=512 -> d_out f32.
// Grid (8, 8) -> 512 waves.
// ---------------------------------------------------------------------------
__global__ void rin_out(const half_t* __restrict__ hA,
                        const float* __restrict__ OP,   // (V, D) f32
                        float* __restrict__ out) {
    __shared__ half_t sA[16 * (DD + PAD)];
    const int m0 = blockIdx.y << 4;
    const int n0 = (blockIdx.x << 7) + ((threadIdx.x >> 5) << 4);

    stage_panel(sA, hA, m0, DD, DD);
    __syncthreads();

    v8f acc = {};
#pragma unroll 4
    for (int k = 0; k < DD; k += 32) {
        __builtin_prefetch(OP + (size_t)n0 * DD + k + 128, 0, 0);
        v16h a = frag_lds(sA, DD + PAD, k);
        v16h b = frag_f32(OP, DD, n0, k);
        acc = WMMA_F16(a, b, acc);
    }

    const int n = n0 + cd_col();
#pragma unroll
    for (int v = 0; v < 8; ++v) {
        const int m = m0 + cd_row(v);
        out[(size_t)m * VV + n] = acc[v];
    }
}

// ---------------------------------------------------------------------------
// Host launch.  Inputs (setup_inputs order):
//  0 input_ids (B,S) i32            1 token_embedding (V,2D) f32
//  2 layers_W (L,N,D) f32           3 layers_bias (L,N) f32
//  4 layers_proj_real (L,D,N) f32   5 layers_proj_imag (L,D,N) f32
//  6 output_proj (V,D) f32          out: (B,V) f32
// ---------------------------------------------------------------------------
extern "C" void kernel_launch(void* const* d_in, const int* in_sizes, int n_in,
                              void* d_out, int out_size, void* d_ws, size_t ws_size,
                              hipStream_t stream) {
    const int*   ids  = (const int*)d_in[0];
    const float* emb  = (const float*)d_in[1];
    const float* W    = (const float*)d_in[2];
    const float* bias = (const float*)d_in[3];
    const float* Pr   = (const float*)d_in[4];
    const float* Pi   = (const float*)d_in[5];
    const float* OP   = (const float*)d_in[6];
    float* out = (float*)d_out;

    // Workspace layout (bytes):
    //   [0, 256K)      h_f : f32 (B*D)
    //   [256K, 384K)   h_h : f16 (B*D)
    //   [384K, 640K)   ct  : f16 (B*N)
    //   [640K, 896K)   st  : f16 (B*N)
    char* ws = (char*)d_ws;
    float*  h_f = (float*)(ws);
    half_t* h_h = (half_t*)(ws + (size_t)BB * DD * 4);
    half_t* ct  = (half_t*)(ws + (size_t)BB * DD * 6);
    half_t* st  = (half_t*)(ws + (size_t)BB * DD * 6 + (size_t)BB * NN * 2);

    const float t_last = (float)(SS - 1) * PHI;

    // 1) 64-step rotation recurrence (elementwise, independent per (b,d)).
    rin_scan<<<(BB * DD) / 256, 256, 0, stream>>>(ids, emb, h_f, h_h);

    // 2) Two resonant layers, applied once (steps 0..62 are dead code).
    for (int l = 0; l < LL; ++l) {
        rin_theta<<<dim3(NN / 128, BB / 16), 256, 0, stream>>>(
            h_h, W + (size_t)l * NN * DD, bias + (size_t)l * NN, ct, st, t_last);
        rin_proj<<<dim3(DD / 128, BB / 16), 256, 0, stream>>>(
            ct, st, Pr + (size_t)l * DD * NN, Pi + (size_t)l * DD * NN, h_f, h_h);
    }

    // 3) Output projection -> logits (B, V).
    rin_out<<<dim3(VV / 128, BB / 16), 256, 0, stream>>>(h_h, OP, out);
}
